// MaskedMHA_73572789780875
// MI455X (gfx1250) — matrix operand order repaired and checked by reading, not compiled, not run on hardware
//
#include <hip/hip_runtime.h>

// ---------------------------------------------------------------------------
// MaskedMHA for MI455X (gfx1250): bf16 WMMA GEMMs with fully async-LDS staged
// A and B (ASYNCcnt-gated ping-pong pipeline) + flash attention, wave32.
// ---------------------------------------------------------------------------

typedef __bf16 bf16;
typedef __attribute__((ext_vector_type(16))) __bf16 v16bf;
typedef __attribute__((ext_vector_type(8)))  __bf16 v8bf;
typedef __attribute__((ext_vector_type(8)))  float  v8f;
typedef __attribute__((ext_vector_type(4)))  float  v4f;

union ABf { v16bf v; v8bf h[2]; };

__device__ __forceinline__ v8f wmma_bf16(v16bf a, v16bf b, v8f c) {
  return __builtin_amdgcn_wmma_f32_16x16x32_bf16(
      false, a, false, b, (short)0, c, false, false);
}

constexpr int BATCH  = 4;
constexpr int SEQ    = 2048;
constexpr int DMODEL = 1024;
constexpr int NH     = 16;
constexpr int DK     = 64;
constexpr int MROWS  = BATCH * SEQ;   // 8192

constexpr int KB    = 32;             // k-block depth
constexpr int NITER = DMODEL / KB;    // 32 (even)
constexpr int TROW  = 40;             // LDS tile row stride in elems (80B)

#define NEG_INF (-__builtin_inff())

// --- CDNA5 async global->LDS copy (ASYNCcnt) -------------------------------
__device__ __forceinline__ void async_b128(unsigned lds_off, const void* g) {
  unsigned long long ga = (unsigned long long)g;
  asm volatile("global_load_async_to_lds_b128 %0, %1, off"
               :: "v"(lds_off), "v"(ga) : "memory");
}
// 5 async ops issued per thread per stage (4 A + 1 B).
#define WAIT_ASYNC_5() asm volatile("s_wait_asynccnt 0x5" ::: "memory")
#define WAIT_ASYNC_0() asm volatile("s_wait_asynccnt 0x0" ::: "memory")

// Stage B slab: WT rows n0..n0+63, cols k0..k0+31. 256 thr x 1 b128.
__device__ __forceinline__ void stage_B(const bf16* __restrict__ wT, int n0,
                                        int k0, unsigned ldsBase, int tid) {
  const int n  = tid >> 2;
  const int ch = tid & 3;
  const bf16* g = wT + (size_t)(n0 + n) * DMODEL + k0 + ch * 8;
  async_b128(ldsBase + (unsigned)(n * TROW + ch * 8) * sizeof(bf16), g);
}

// Stage A slab: rows mBase..mBase+255, cols k0..k0+31. 256 thr x 4 b128.
__device__ __forceinline__ void stage_A(const bf16* __restrict__ A, int mBase,
                                        int k0, unsigned ldsBase, int tid) {
  const int rr = tid >> 2;
  const int ch = tid & 3;
#pragma unroll
  for (int i = 0; i < 4; ++i) {
    const int row = rr + 64 * i;
    const bf16* g = A + (size_t)(mBase + row) * DMODEL + k0 + ch * 8;
    async_b128(ldsBase + (unsigned)(row * TROW + ch * 8) * sizeof(bf16), g);
  }
}

// ---------------------------------------------------------------------------
// Stage 0: conversions
// ---------------------------------------------------------------------------
__global__ __launch_bounds__(256) void k_cvt_x(const float* __restrict__ x,
                                               bf16* __restrict__ xb) {
  const int i = (blockIdx.x * blockDim.x + threadIdx.x) * 8;
  v4f a = *(const v4f*)(x + i);
  v4f b = *(const v4f*)(x + i + 4);
  v8bf o;
  o[0] = (bf16)a[0]; o[1] = (bf16)a[1]; o[2] = (bf16)a[2]; o[3] = (bf16)a[3];
  o[4] = (bf16)b[0]; o[5] = (bf16)b[1]; o[6] = (bf16)b[2]; o[7] = (bf16)b[3];
  *(v8bf*)(xb + i) = o;
}

__global__ __launch_bounds__(256) void k_cvt_wT(const float* __restrict__ w,
                                                bf16* __restrict__ wT) {
  const int i = blockIdx.x * blockDim.x + threadIdx.x;   // 0 .. 1M
  const int k = i & (DMODEL - 1);
  const int n = i >> 10;
  wT[i] = (bf16)w[k * DMODEL + n];
}

// ---------------------------------------------------------------------------
// GEMM core: 8 waves; wave tile 32x64; WG tile 256x64. A and B both staged in
// double-buffered LDS via async copies; only ASYNCcnt/DScnt waits in-loop.
// ---------------------------------------------------------------------------
#define STAGE(APTR, WTPTR, K, BUF)                                             \
  stage_A((APTR), mBase, (K), aBase##BUF, tid);                                \
  stage_B((WTPTR), n0, (K), bBase##BUF, tid);

#define COMPUTE(BUF)                                                           \
  {                                                                            \
    ABf a0, a1;                                                                \
    const bf16* ar0 = &As[(BUF)][(mW + ln) * TROW];                            \
    const bf16* ar1 = &As[(BUF)][(mW + 16 + ln) * TROW];                       \
    a0.h[0] = *(const v8bf*)(ar0 + aoff);                                      \
    a0.h[1] = *(const v8bf*)(ar0 + 16 + aoff);                                 \
    a1.h[0] = *(const v8bf*)(ar1 + aoff);                                      \
    a1.h[1] = *(const v8bf*)(ar1 + 16 + aoff);                                 \
    _Pragma("unroll")                                                          \
    for (int j = 0; j < 4; ++j) {                                              \
      const bf16* brow = &Bs[(BUF)][(16 * j + ln) * TROW + boff];              \
      ABf bm;                                                                  \
      bm.h[0] = *(const v8bf*)(brow);                                          \
      bm.h[1] = *(const v8bf*)(brow + 16);                                     \
      acc[0][j] = wmma_bf16(a0.v, bm.v, acc[0][j]);                            \
      acc[1][j] = wmma_bf16(a1.v, bm.v, acc[1][j]);                            \
    }                                                                          \
  }

#define GEMM_CORE(APTR, WTPTR)                                                 \
  __shared__ bf16 As[2][256 * TROW];   /* 2 x 20 KB */                         \
  __shared__ bf16 Bs[2][64 * TROW];    /* 2 x  5 KB */                         \
  const int tid   = threadIdx.x;                                               \
  const int lane  = tid & 31;                                                  \
  const int hi    = lane >> 4;                                                 \
  const int ln    = lane & 15;                                                 \
  const int aoff  = hi * 8;                                                    \
  const int boff  = hi * 8;                                                    \
  const int mW    = (tid >> 5) * 32;                                           \
  const int m0    = mBase + mW;                                                \
  const unsigned aBase0 = (unsigned)(uintptr_t)&As[0][0];                      \
  const unsigned aBase1 = (unsigned)(uintptr_t)&As[1][0];                      \
  const unsigned bBase0 = (unsigned)(uintptr_t)&Bs[0][0];                      \
  const unsigned bBase1 = (unsigned)(uintptr_t)&Bs[1][0];                      \
  v8f acc[2][4] = {};                                                          \
  STAGE(APTR, WTPTR, 0, 0)                                                     \
  for (int it = 0; it < NITER; it += 2) {                                      \
    /* phase 0: prefetch k=it+1 into buf1; consume buf0 */                     \
    STAGE(APTR, WTPTR, (it + 1) * KB, 1)                                       \
    WAIT_ASYNC_5();                                                            \
    __syncthreads();                                                           \
    COMPUTE(0)                                                                 \
    __syncthreads();                                                           \
    /* phase 1: prefetch k=it+2 into buf0; consume buf1 */                     \
    if (it + 2 < NITER) {                                                      \
      STAGE(APTR, WTPTR, (it + 2) * KB, 0)                                     \
      WAIT_ASYNC_5();                                                          \
    } else {                                                                   \
      WAIT_ASYNC_0();                                                          \
    }                                                                          \
    __syncthreads();                                                           \
    COMPUTE(1)                                                                 \
    __syncthreads();                                                           \
  }

// ---------------------------------------------------------------------------
// Stage 1: fused QKV projection. gridDim = (M/256, D/64, 3).
// ---------------------------------------------------------------------------
__global__ __launch_bounds__(256)
void k_qkv(const bf16* __restrict__ xb,
           const bf16* __restrict__ wqT, const bf16* __restrict__ wkT,
           const bf16* __restrict__ wvT,
           const float* __restrict__ bq, const float* __restrict__ bk,
           const float* __restrict__ bv,
           bf16* __restrict__ Qb, bf16* __restrict__ Kb, bf16* __restrict__ Vb) {
  const int sel = blockIdx.z;
  const bf16*  wT   = (sel == 0) ? wqT : ((sel == 1) ? wkT : wvT);
  const float* bias = (sel == 0) ? bq  : ((sel == 1) ? bk  : bv);
  const int mBase = blockIdx.x * 256;
  const int n0    = blockIdx.y * 64;

  GEMM_CORE(xb, wT)

#pragma unroll
  for (int ms = 0; ms < 2; ++ms) {
#pragma unroll
    for (int j = 0; j < 4; ++j) {
      const int c = n0 + 16 * j + ln;
      const int h = c >> 6, d = c & 63;
      const float bs = bias[c];
#pragma unroll
      for (int v = 0; v < 8; ++v) {
        const int r  = m0 + 16 * ms + v + hi * 8;
        const int b_ = r >> 11;
        const int s  = r & (SEQ - 1);
        const float val = acc[ms][j][v] + bs;
        if (sel == 2) {   // V stored d-major: [b][h][d][S]
          Vb[(((size_t)b_ * NH + h) * DK + d) * SEQ + s] = (bf16)val;
        } else {          // Q/K stored [b][h][s][d]
          bf16* dst = (sel == 0) ? Qb : Kb;
          dst[(((size_t)b_ * NH + h) * SEQ + s) * DK + d] = (bf16)val;
        }
      }
    }
  }
}

// ---------------------------------------------------------------------------
// Stage 2: causal flash attention. gridDim = (S/128, H, B); 8 waves / block.
// ---------------------------------------------------------------------------
__global__ __launch_bounds__(256)
void k_flash(const bf16* __restrict__ Qb, const bf16* __restrict__ Kb,
             const bf16* __restrict__ Vb, bf16* __restrict__ Ob) {
  __shared__ bf16 Pl[8][16 * 40];

  const int lane = threadIdx.x & 31;
  const int wave = threadIdx.x >> 5;
  const int h    = blockIdx.y;
  const int b    = blockIdx.z;
  const int q0   = blockIdx.x * 128 + wave * 16;
  const int hi   = lane >> 4, ln = lane & 15;
  const int aoff = hi * 8, boff = hi * 8;

  const size_t headQK = ((size_t)b * NH + h) * SEQ;
  const size_t headV  = ((size_t)b * NH + h) * DK;

  ABf qa[2];
  const bf16* qrow = Qb + (headQK + q0 + ln) * DK;
#pragma unroll
  for (int t = 0; t < 2; ++t) {
    qa[t].h[0] = *(const v8bf*)(qrow + 32 * t + aoff);
    qa[t].h[1] = *(const v8bf*)(qrow + 32 * t + 16 + aoff);
  }

  v8f o[4] = {};
  float mi[8], li[8];
#pragma unroll
  for (int v = 0; v < 8; ++v) { mi[v] = NEG_INF; li[v] = 0.f; }

  const float scale = 0.125f;
  const int tend = q0 + 16;

  for (int t0 = 0; t0 < tend; t0 += 32) {
    v8f sc[2];
#pragma unroll
    for (int j = 0; j < 2; ++j) {
      const bf16* krow = Kb + (headQK + t0 + 16 * j + ln) * DK;
      ABf kb0, kb1;
      kb0.h[0] = *(const v8bf*)(krow + boff);
      kb0.h[1] = *(const v8bf*)(krow + boff + 16);
      kb1.h[0] = *(const v8bf*)(krow + 32 + boff);
      kb1.h[1] = *(const v8bf*)(krow + 32 + boff + 16);
      v8f z = {};
      z = wmma_bf16(qa[0].v, kb0.v, z);
      z = wmma_bf16(qa[1].v, kb1.v, z);
      sc[j] = z;
    }

#pragma unroll
    for (int v = 0; v < 8; ++v) {
      const int rq = q0 + v + hi * 8;
#pragma unroll
      for (int j = 0; j < 2; ++j) {
        const int tk = t0 + 16 * j + ln;
        const float xs = sc[j][v] * scale;
        sc[j][v] = (tk <= rq) ? xs : NEG_INF;
      }
      float mx = fmaxf(sc[0][v], sc[1][v]);
      mx = fmaxf(mx, __shfl_xor(mx, 1, 32));
      mx = fmaxf(mx, __shfl_xor(mx, 2, 32));
      mx = fmaxf(mx, __shfl_xor(mx, 4, 32));
      mx = fmaxf(mx, __shfl_xor(mx, 8, 32));
      const float newm  = fmaxf(mi[v], mx);
      const float alpha = __expf(mi[v] - newm);
      float rs = 0.f;
#pragma unroll
      for (int j = 0; j < 2; ++j) {
        const float e = __expf(sc[j][v] - newm);
        sc[j][v] = e;
        rs += e;
      }
      rs += __shfl_xor(rs, 1, 32);
      rs += __shfl_xor(rs, 2, 32);
      rs += __shfl_xor(rs, 4, 32);
      rs += __shfl_xor(rs, 8, 32);
      li[v] = li[v] * alpha + rs;
      mi[v] = newm;
#pragma unroll
      for (int jd = 0; jd < 4; ++jd) o[jd][v] *= alpha;
    }

    asm volatile("" ::: "memory");
#pragma unroll
    for (int v = 0; v < 8; ++v) {
      const int row = v + hi * 8;
      Pl[wave][row * 40 + ln]      = (bf16)sc[0][v];
      Pl[wave][row * 40 + 16 + ln] = (bf16)sc[1][v];
    }
    asm volatile("" ::: "memory");
    ABf pa;
    const bf16* prow = &Pl[wave][ln * 40];
    pa.h[0] = *(const v8bf*)(prow + aoff);
    pa.h[1] = *(const v8bf*)(prow + 16 + aoff);

#pragma unroll
    for (int jd = 0; jd < 4; ++jd) {
      const bf16* vrow = Vb + (headV + 16 * jd + ln) * SEQ + t0;
      ABf vb;
      vb.h[0] = *(const v8bf*)(vrow + boff);
      vb.h[1] = *(const v8bf*)(vrow + boff + 16);
      o[jd] = wmma_bf16(pa.v, vb.v, o[jd]);
    }
  }

#pragma unroll
  for (int jd = 0; jd < 4; ++jd) {
#pragma unroll
    for (int v = 0; v < 8; ++v) {
      const int s = q0 + v + hi * 8;
      const int d = 16 * jd + ln;
      const float val = o[jd][v] / li[v];
      Ob[(((size_t)b * SEQ + s) * NH + h) * DK + d] = (bf16)val;
    }
  }
}

// ---------------------------------------------------------------------------
// Stage 3: output projection. gridDim = (M/256, D/64).
// ---------------------------------------------------------------------------
__global__ __launch_bounds__(256)
void k_outproj(const bf16* __restrict__ Ab, const bf16* __restrict__ woT,
               const float* __restrict__ bo, float* __restrict__ out) {
  const int mBase = blockIdx.x * 256;
  const int n0    = blockIdx.y * 64;

  GEMM_CORE(Ab, woT)

#pragma unroll
  for (int ms = 0; ms < 2; ++ms) {
#pragma unroll
    for (int j = 0; j < 4; ++j) {
      const int c = n0 + 16 * j + ln;
      const float bs = bo[c];
#pragma unroll
      for (int v = 0; v < 8; ++v) {
        const int r = m0 + 16 * ms + v + hi * 8;
        out[(size_t)r * DMODEL + c] = acc[ms][j][v] + bs;
      }
    }
  }
}

// ---------------------------------------------------------------------------
extern "C" void kernel_launch(void* const* d_in, const int* in_sizes, int n_in,
                              void* d_out, int out_size, void* d_ws, size_t ws_size,
                              hipStream_t stream) {
  const float* x  = (const float*)d_in[0];
  const float* wq = (const float*)d_in[1];
  const float* bq = (const float*)d_in[2];
  const float* wk = (const float*)d_in[3];
  const float* bk = (const float*)d_in[4];
  const float* wv = (const float*)d_in[5];
  const float* bv = (const float*)d_in[6];
  const float* wo = (const float*)d_in[7];
  const float* bo = (const float*)d_in[8];
  float* out = (float*)d_out;

  char* p = (char*)d_ws;
  size_t off = 0;
  auto carve = [&](size_t bytes) { char* q = p + off; off += bytes; return q; };

  bf16* xb  = (bf16*)carve((size_t)MROWS * DMODEL * sizeof(bf16));
  bf16* wqT = (bf16*)carve((size_t)DMODEL * DMODEL * sizeof(bf16));
  bf16* wkT = (bf16*)carve((size_t)DMODEL * DMODEL * sizeof(bf16));
  bf16* wvT = (bf16*)carve((size_t)DMODEL * DMODEL * sizeof(bf16));
  bf16* woT = (bf16*)carve((size_t)DMODEL * DMODEL * sizeof(bf16));
  bf16* Qb  = (bf16*)carve((size_t)MROWS * DMODEL * sizeof(bf16));
  bf16* Kb  = (bf16*)carve((size_t)MROWS * DMODEL * sizeof(bf16));
  bf16* Vb  = (bf16*)carve((size_t)MROWS * DMODEL * sizeof(bf16));
  bf16* Ob  = (bf16*)carve((size_t)MROWS * DMODEL * sizeof(bf16));
  (void)ws_size; (void)in_sizes; (void)n_in; (void)out_size;

  k_cvt_x<<<(MROWS * DMODEL) / 8 / 256, 256, 0, stream>>>(x, xb);
  const int wblocks = (DMODEL * DMODEL) / 256;
  k_cvt_wT<<<wblocks, 256, 0, stream>>>(wq, wqT);
  k_cvt_wT<<<wblocks, 256, 0, stream>>>(wk, wkT);
  k_cvt_wT<<<wblocks, 256, 0, stream>>>(wv, wvT);
  k_cvt_wT<<<wblocks, 256, 0, stream>>>(wo, woT);

  k_qkv<<<dim3(MROWS / 256, DMODEL / 64, 3), 256, 0, stream>>>(
      xb, wqT, wkT, wvT, bq, bk, bv, Qb, Kb, Vb);

  k_flash<<<dim3(SEQ / 128, NH, BATCH), 256, 0, stream>>>(Qb, Kb, Vb, Ob);

  k_outproj<<<dim3(MROWS / 256, DMODEL / 64), 256, 0, stream>>>(Ob, woT, bo, out);
}
